// GATModel_60129542733
// MI455X (gfx1250) — matrix operand order repaired and checked by reading, not compiled, not run on hardware
//
#include <hip/hip_runtime.h>
#include <hip/hip_bf16.h>

// ---------------------------------------------------------------------------
// 2-layer GAT on gfx1250 (MI455X).
//  * Dense projections (x@W1, h@W2) run on v_wmma_f32_16x16x32_f16 (f16 in,
//    f32 accumulate). Inputs are ~N(0,0.05^2)-scaled so f16 operands are fine.
//  * Graph softmax/aggregation: 3 edge passes with fp32 atomics (max encoded
//    as order-preserving uint). This is the memory-bound part (~1.7GB random
//    L2-resident traffic); WMMA covers the matrix ops.
// ---------------------------------------------------------------------------

#define IN_CH   128
#define C_TOT   64          // HEADS*HID == OUT == 64
#define HEADS1  8
#define NEG_SLOPE 0.2f

typedef __attribute__((ext_vector_type(16))) _Float16 v16h;
typedef __attribute__((ext_vector_type(8)))  _Float16 v8h;
typedef __attribute__((ext_vector_type(8)))  float    v8f;

union V16 { v16h v; v8h h[2]; };

static __device__ __forceinline__ unsigned fenc(float f) {
  unsigned u = __float_as_uint(f);
  return (u & 0x80000000u) ? ~u : (u | 0x80000000u);
}
static __device__ __forceinline__ float fdec(unsigned u) {
  return __uint_as_float((u & 0x80000000u) ? (u ^ 0x80000000u) : ~u);
}
static __device__ __forceinline__ void edge_nodes(const int* __restrict__ ei,
                                                  int e, int E, int& s, int& d) {
  if (e < E) { s = ei[e]; d = ei[E + e]; } else { s = e - E; d = e - E; }
}

// ---------------------------------------------------------------------------
// Small utility kernels
// ---------------------------------------------------------------------------
__global__ void k_f32_to_f16(const float* __restrict__ in,
                             _Float16* __restrict__ out, int n) {
  int i = blockIdx.x * blockDim.x + threadIdx.x;
  if (i < n) out[i] = (_Float16)in[i];
}

__global__ void k_fill_u32(unsigned* __restrict__ p, unsigned v, int n) {
  int i = blockIdx.x * blockDim.x + threadIdx.x;
  if (i < n) p[i] = v;
}

// Pack W (K x 64 row-major, f32) into WMMA B-operand lane layout (f16):
// Wp[((ct*ksteps + ks)*32 + lane)*16 + j] = W[ks*32 + (lane<16?0:8) + (j&7) + (j&8?16:0)][ct*16 + (lane&15)]
__global__ void k_pack_w(const float* __restrict__ W, _Float16* __restrict__ Wp,
                         int K) {
  int ksteps = K >> 5;
  int total  = 4 * ksteps * 32 * 16;
  int idx = blockIdx.x * blockDim.x + threadIdx.x;
  if (idx >= total) return;
  int j    = idx & 15;
  int lane = (idx >> 4) & 31;
  int ks   = (idx >> 9) % ksteps;
  int ct   = (idx >> 9) / ksteps;
  int krow = ks * 32 + ((lane < 16) ? 0 : 8) + (j & 7) + ((j & 8) ? 16 : 0);
  int col  = ct * 16 + (lane & 15);
  Wp[idx] = (_Float16)W[(size_t)krow * C_TOT + col];
}

// ---------------------------------------------------------------------------
// WMMA GEMM: out[nrows x 64] = A[nrows x K] (f16) * Wpacked (f16), f32 accum.
// One wave per 16-row tile; 4 column tiles (64 cols); K in steps of 32.
// A layout per ISA 7.12.2 (16-bit A 16x32): lane<16 -> row=lane, K 0-7 & 16-23;
// lane>=16 -> row=lane-16, K 8-15 & 24-31. Each half is 8 contiguous halves.
// ---------------------------------------------------------------------------
__global__ __launch_bounds__(256) void k_gemm_wmma(
    const _Float16* __restrict__ A, const _Float16* __restrict__ Bp,
    float* __restrict__ out, int nrows, int K) {
  const int ksteps = K >> 5;
  const int lane = threadIdx.x & 31;
  const int wave = threadIdx.x >> 5;
  const int tile = blockIdx.x * 8 + wave;
  if (tile * 16 >= nrows) return;
  const int m0 = tile * 16;
  const size_t row = (size_t)(m0 + (lane & 15));
  const int kb = (lane < 16) ? 0 : 8;

  v8f acc[4] = {v8f{}, v8f{}, v8f{}, v8f{}};
  for (int ks = 0; ks < ksteps; ++ks) {
    V16 a;
    const _Float16* ap = A + row * K + ks * 32 + kb;
    a.h[0] = *(const v8h*)(ap);        // K = kb .. kb+7
    a.h[1] = *(const v8h*)(ap + 16);   // K = kb+16 .. kb+23
#pragma unroll
    for (int ct = 0; ct < 4; ++ct) {
      V16 b;
      const _Float16* bp = Bp + (size_t)((ct * ksteps + ks) * 32 + lane) * 16;
      b.h[0] = *(const v8h*)(bp);
      b.h[1] = *(const v8h*)(bp + 8);
      acc[ct] = __builtin_amdgcn_wmma_f32_16x16x32_f16(
          false, a.v, false, b.v, (short)0, acc[ct], false, false);
    }
  }
  // D layout: VGPR r -> M = r (+8 for lanes 16-31), N = lane&15
  const int rbase = (lane >> 4) << 3;
  const int col = lane & 15;
#pragma unroll
  for (int ct = 0; ct < 4; ++ct)
#pragma unroll
    for (int r = 0; r < 8; ++r)
      out[(size_t)(m0 + rbase + r) * C_TOT + ct * 16 + col] = acc[ct][r];
}

// ---------------------------------------------------------------------------
// Per-node attention logits: als[n,h] = sum_c h[n, h*C+c] * a_s[h*C+c]
// (H=8,C=8 for layer 1; H=1,C=64 for layer 2)
// ---------------------------------------------------------------------------
__global__ void k_node_logits(const float* __restrict__ hmat,
                              const float* __restrict__ a_s,
                              const float* __restrict__ a_d,
                              float* __restrict__ als, float* __restrict__ ald,
                              int N, int H) {
  int idx = blockIdx.x * blockDim.x + threadIdx.x;
  if (idx >= N * H) return;
  int n = idx / H, hd = idx % H;
  int C = C_TOT / H;
  const float* row = hmat + (size_t)n * C_TOT + hd * C;
  float s = 0.f, d = 0.f;
  for (int c = 0; c < C; ++c) {
    float v = row[c];
    s += v * a_s[hd * C + c];
    d += v * a_d[hd * C + c];
  }
  als[idx] = s;
  ald[idx] = d;
}

// ---------------------------------------------------------------------------
// Edge passes (segment softmax + aggregation). Edges e<E come from edge_index,
// e>=E are self-loops on node e-E.
// ---------------------------------------------------------------------------
__global__ void k_edge_max(const int* __restrict__ ei,
                           const float* __restrict__ als,
                           const float* __restrict__ ald,
                           unsigned* __restrict__ emax, int E, int N, int H) {
  int idx = blockIdx.x * blockDim.x + threadIdx.x;
  if (idx >= (E + N) * H) return;
  int e = idx / H, hd = idx % H;
  int s, d; edge_nodes(ei, e, E, s, d);
  float l = als[s * H + hd] + ald[d * H + hd];
  l = (l > 0.f) ? l : NEG_SLOPE * l;
  atomicMax(&emax[d * H + hd], fenc(l));
}

__global__ void k_edge_sum(const int* __restrict__ ei,
                           const float* __restrict__ als,
                           const float* __restrict__ ald,
                           const unsigned* __restrict__ emax,
                           float* __restrict__ denom, int E, int N, int H) {
  int idx = blockIdx.x * blockDim.x + threadIdx.x;
  if (idx >= (E + N) * H) return;
  int e = idx / H, hd = idx % H;
  int s, d; edge_nodes(ei, e, E, s, d);
  float l = als[s * H + hd] + ald[d * H + hd];
  l = (l > 0.f) ? l : NEG_SLOPE * l;
  float ex = __expf(l - fdec(emax[d * H + hd]));
  atomicAdd(&denom[d * H + hd], ex);
}

// One thread per (edge, slot); each slot covers 8 of the 64 channels.
// head = slot & (H-1) (H is 8 or 1, both powers of two).
__global__ void k_edge_msg(const int* __restrict__ ei,
                           const float* __restrict__ als,
                           const float* __restrict__ ald,
                           const unsigned* __restrict__ emax,
                           const float* __restrict__ denom,
                           const float* __restrict__ hmat,
                           float* __restrict__ agg, int E, int N, int H) {
  int idx = blockIdx.x * blockDim.x + threadIdx.x;
  if (idx >= (E + N) * 8) return;
  int e = idx >> 3, slot = idx & 7;
  int hd = slot & (H - 1);
  int s, d; edge_nodes(ei, e, E, s, d);
  float l = als[s * H + hd] + ald[d * H + hd];
  l = (l > 0.f) ? l : NEG_SLOPE * l;
  float ex = __expf(l - fdec(emax[d * H + hd]));
  float alpha = ex / (denom[d * H + hd] + 1e-16f);
  const float* hs = hmat + (size_t)s * C_TOT + slot * 8;
  float* ag = agg + (size_t)d * C_TOT + slot * 8;
#pragma unroll
  for (int c = 0; c < 8; ++c) atomicAdd(&ag[c], hs[c] * alpha);
}

// out1 = elu(agg + b1), stored as f16 (feeds layer-2 WMMA GEMM)
__global__ void k_bias_elu_f16(const float* __restrict__ agg,
                               const float* __restrict__ b,
                               _Float16* __restrict__ outh, int total) {
  int i = blockIdx.x * blockDim.x + threadIdx.x;
  if (i >= total) return;
  float v = agg[i] + b[i & (C_TOT - 1)];
  float e = (v > 0.f) ? v : (__expf(v) - 1.f);
  outh[i] = (_Float16)e;
}

__global__ void k_bias_out(const float* __restrict__ agg,
                           const float* __restrict__ b,
                           float* __restrict__ out, int total) {
  int i = blockIdx.x * blockDim.x + threadIdx.x;
  if (i >= total) return;
  out[i] = agg[i] + b[i & (C_TOT - 1)];
}

// ---------------------------------------------------------------------------
static inline int gdiv(long long a, int b) { return (int)((a + b - 1) / b); }

extern "C" void kernel_launch(void* const* d_in, const int* in_sizes, int n_in,
                              void* d_out, int out_size, void* d_ws, size_t ws_size,
                              hipStream_t stream) {
  const float* x      = (const float*)d_in[0];
  const int*   ei     = (const int*)  d_in[1];
  const float* W1     = (const float*)d_in[2];
  const float* a_src1 = (const float*)d_in[3];
  const float* a_dst1 = (const float*)d_in[4];
  const float* b1     = (const float*)d_in[5];
  const float* W2     = (const float*)d_in[6];
  const float* a_src2 = (const float*)d_in[7];
  const float* a_dst2 = (const float*)d_in[8];
  const float* b2     = (const float*)d_in[9];
  float* out = (float*)d_out;

  const int N = in_sizes[0] / IN_CH;   // 100000
  const int E = in_sizes[1] / 2;       // 1600000
  const int ET = E + N;                // edges incl. self-loops

  // ---- workspace carve-up (256B aligned) ----
  char* w = (char*)d_ws;
  auto take = [&](size_t bytes) -> char* {
    char* p = w; w += (bytes + 255) & ~(size_t)255; return p;
  };
  _Float16* xh     = (_Float16*)take((size_t)N * IN_CH * 2);  // dead after GEMM1
  _Float16* W1p    = (_Float16*)take(4 * 4 * 32 * 16 * 2);
  _Float16* W2p    = (_Float16*)take(4 * 2 * 32 * 16 * 2);
  float*    h1     = (float*)   take((size_t)N * C_TOT * 4);
  float*    als1   = (float*)   take((size_t)N * HEADS1 * 4);
  float*    ald1   = (float*)   take((size_t)N * HEADS1 * 4);
  unsigned* emax1  = (unsigned*)take((size_t)N * HEADS1 * 4);
  float*    denom1 = (float*)   take((size_t)N * HEADS1 * 4);
  float*    agg1   = (float*)   take((size_t)N * C_TOT * 4);
  _Float16* h1h    = (_Float16*)take((size_t)N * C_TOT * 2);
  float*    h2     = (float*)   take((size_t)N * C_TOT * 4);
  float*    als2   = (float*)   take((size_t)N * 4);
  float*    ald2   = (float*)   take((size_t)N * 4);
  unsigned* emax2  = (unsigned*)take((size_t)N * 4);
  float*    denom2 = (float*)   take((size_t)N * 4);
  float*    agg2   = (float*)xh;  // alias: xh dead after GEMM1, zeroed post-GEMM1

  const int B = 256;
  // Order-preserving encode of -inf (0xFF800000): ~u == 0x007FFFFF.
  const unsigned NEG_INF_ENC = 0x007FFFFFu;

  // ---- layer 1 ----
  k_f32_to_f16<<<gdiv((long long)N * IN_CH, B), B, 0, stream>>>(x, xh, N * IN_CH);
  k_pack_w<<<gdiv(4 * 4 * 32 * 16, B), B, 0, stream>>>(W1, W1p, IN_CH);
  k_pack_w<<<gdiv(4 * 2 * 32 * 16, B), B, 0, stream>>>(W2, W2p, C_TOT);

  (void)hipMemsetAsync(agg1,   0, (size_t)N * C_TOT * 4, stream);
  (void)hipMemsetAsync(denom1, 0, (size_t)N * HEADS1 * 4, stream);
  k_fill_u32<<<gdiv(N * HEADS1, B), B, 0, stream>>>(emax1, NEG_INF_ENC, N * HEADS1);

  k_gemm_wmma<<<gdiv(N / 16, 8), B, 0, stream>>>(xh, W1p, h1, N, IN_CH);
  k_node_logits<<<gdiv((long long)N * HEADS1, B), B, 0, stream>>>(
      h1, a_src1, a_dst1, als1, ald1, N, HEADS1);

  k_edge_max<<<gdiv((long long)ET * HEADS1, B), B, 0, stream>>>(
      ei, als1, ald1, emax1, E, N, HEADS1);
  k_edge_sum<<<gdiv((long long)ET * HEADS1, B), B, 0, stream>>>(
      ei, als1, ald1, emax1, denom1, E, N, HEADS1);
  k_edge_msg<<<gdiv((long long)ET * 8, B), B, 0, stream>>>(
      ei, als1, ald1, emax1, denom1, h1, agg1, E, N, HEADS1);

  k_bias_elu_f16<<<gdiv((long long)N * C_TOT, B), B, 0, stream>>>(
      agg1, b1, h1h, N * C_TOT);

  // ---- layer 2 ---- (agg2 aliases xh; safe to zero only now)
  (void)hipMemsetAsync(agg2,   0, (size_t)N * C_TOT * 4, stream);
  (void)hipMemsetAsync(denom2, 0, (size_t)N * 4, stream);
  k_fill_u32<<<gdiv(N, B), B, 0, stream>>>(emax2, NEG_INF_ENC, N);

  k_gemm_wmma<<<gdiv(N / 16, 8), B, 0, stream>>>(h1h, W2p, h2, N, C_TOT);
  k_node_logits<<<gdiv(N, B), B, 0, stream>>>(h2, a_src2, a_dst2, als2, ald2, N, 1);

  k_edge_max<<<gdiv(ET, B), B, 0, stream>>>(ei, als2, ald2, emax2, E, N, 1);
  k_edge_sum<<<gdiv(ET, B), B, 0, stream>>>(ei, als2, ald2, emax2, denom2, E, N, 1);
  k_edge_msg<<<gdiv((long long)ET * 8, B), B, 0, stream>>>(
      ei, als2, ald2, emax2, denom2, h2, agg2, E, N, 1);

  k_bias_out<<<gdiv((long long)N * C_TOT, B), B, 0, stream>>>(
      agg2, b2, out, N * C_TOT);
  (void)n_in; (void)out_size; (void)ws_size;
}
